// BiSeparableConv_79035988181216
// MI455X (gfx1250) — matrix operand order repaired and checked by reading, not compiled, not run on hardware
//
#include <hip/hip_runtime.h>

// ---------------------------------------------------------------------------
// BiSeparableConv (ternary depthwise 3x3 + ternary pointwise 1x1), fused.
//
// MI455X reasoning: HBM-bound problem (154 MB traffic @ 23.3 TB/s ~ 6.6us vs
// ~3us of f16-WMMA GEMM work), so we read x exactly once, keep the depthwise
// intermediate in LDS (f16), and feed v_wmma_f32_16x16x32_f16 directly from
// LDS/L2. Ternary weights are exact in f16; accumulation is f32.
// ---------------------------------------------------------------------------

typedef __attribute__((ext_vector_type(16))) _Float16 v16h;
typedef __attribute__((ext_vector_type(8)))  _Float16 v8h;
typedef __attribute__((ext_vector_type(8)))  float    v8f;

namespace {
constexpr float EPSQ = 1e-5f;
constexpr int NB = 16, C = 384, H = 56, W = 56;
constexpr int HW  = H * W;
constexpr int WT  = 4;          // ceil(56/16) W-tiles per row
constexpr int KT  = C / 32;     // 12 K-iterations of depth 32
constexpr int YSTR = 392;       // padded LDS row stride in f16 (bank-conflict free)

// workspace layout (bytes)
constexpr size_t WS_MDW = 0;                      // float[384]   per-channel dw scale
constexpr size_t WS_UDW = 1536;                   // float[3456]  ternary dw weights
constexpr size_t WS_MPW = WS_UDW + 3456 * 4;      // float[1]     global pw scale (15360)
constexpr size_t WS_WQ  = 15488;                  // f16[147456]  ternary pw weights (16B aligned)
} // namespace

// --- per-channel ternary quantization of the 3x3 depthwise weights ----------
__global__ void k_quant_dw(const float* __restrict__ w_dw,
                           float* __restrict__ mdw, float* __restrict__ udw) {
    int c = threadIdx.x;
    if (c >= C) return;
    float v[9];
    float s = 0.f;
    #pragma unroll
    for (int j = 0; j < 9; ++j) { v[j] = w_dw[c * 9 + j]; s += fabsf(v[j]); }
    float m = fmaxf(s * (1.f / 9.f), EPSQ);
    mdw[c] = m;                       // y is multiplied by m (the 1/s_dw)
    float inv = 1.f / m;
    #pragma unroll
    for (int j = 0; j < 9; ++j) {
        float u = rintf(v[j] * inv);
        udw[c * 9 + j] = fminf(1.f, fmaxf(-1.f, u));
    }
}

// --- global mean(|w_pw|) reduction ------------------------------------------
__global__ void k_pw_scale(const float* __restrict__ w_pw, float* __restrict__ mpw) {
    __shared__ float red[256];
    float s = 0.f;
    for (int i = threadIdx.x; i < C * C; i += 256) s += fabsf(w_pw[i]);
    red[threadIdx.x] = s;
    __syncthreads();
    for (int off = 128; off > 0; off >>= 1) {
        if ((int)threadIdx.x < off) red[threadIdx.x] += red[threadIdx.x + off];
        __syncthreads();
    }
    if (threadIdx.x == 0) mpw[0] = fmaxf(red[0] * (1.f / (C * C)), EPSQ);
}

// --- ternarize pointwise weights into f16 (wq[o][k], L2-resident) -----------
__global__ void k_quant_pw(const float* __restrict__ w_pw,
                           const float* __restrict__ mpw,
                           _Float16* __restrict__ wq) {
    int i = blockIdx.x * 256 + threadIdx.x;
    if (i >= C * C) return;
    float inv = 1.f / mpw[0];
    float u = rintf(w_pw[i] * inv);
    wq[i] = (_Float16)fminf(1.f, fmaxf(-1.f, u));
}

// --- fused depthwise(3x3) -> LDS -> pointwise(1x1) via WMMA -----------------
__global__ void __launch_bounds__(256)
k_fused(const float* __restrict__ x,
        const float* __restrict__ mdw,
        const float* __restrict__ udw,
        const float* __restrict__ mpw,
        const _Float16* __restrict__ wq,
        float* __restrict__ out) {
    __shared__ _Float16 yT[16 * YSTR];   // y[m][k] (f16), padded stride

    const int tile = blockIdx.x;
    const int tw = tile % WT;
    const int h  = (tile / WT) % H;
    const int nb = tile / (WT * H);
    const int w0 = tw * 16;

    // Phase A: depthwise 3x3 ternary conv for 384 channels x 16 W-positions.
    // 24 outputs per thread; fold the per-channel scale m_c into y here.
    for (int idx = threadIdx.x; idx < C * 16; idx += 256) {
        const int c = idx >> 4;
        const int m = idx & 15;
        const int w = w0 + m;
        float acc = 0.f;
        if (w < W) {
            const float* xc = x + (size_t)(nb * C + c) * HW;
            const float* uw = udw + c * 9;
            #pragma unroll
            for (int dh = -1; dh <= 1; ++dh) {
                const int hh = h + dh;
                if (hh < 0 || hh >= H) continue;
                const float* xr = xc + hh * W;
                #pragma unroll
                for (int dw = -1; dw <= 1; ++dw) {
                    const int ww = w + dw;
                    if (ww < 0 || ww >= W) continue;
                    acc = fmaf(xr[ww], uw[(dh + 1) * 3 + (dw + 1)], acc);
                }
            }
            acc *= mdw[c];
        }
        yT[m * YSTR + c] = (_Float16)acc;
    }
    __syncthreads();

    const float spw = mpw[0];
    const int lane = threadIdx.x & 31;
    const int wave = threadIdx.x >> 5;
    const int lm   = lane & 15;          // A: row M   | B: col N | C: col N
    const int kh   = lane >> 4;          // K-half select

    // Phase B: 8 waves x 3 tiles = 24 output-channel tiles of 16.
    for (int t = 0; t < 3; ++t) {
        const int ntile = wave * 3 + t;
        const int o = ntile * 16 + lm;   // this lane's output channel (B col)
        v8f acc = {};
        for (int kk = 0; kk < KT; ++kk) {
            // A-fragment (16x32 f16): lane lm holds row M=lm; VGPR0-3 hold
            // K = kh*8 + 0..7, VGPR4-7 hold K = kh*8 + 16..23 (ISA layout).
            const int abase = lm * YSTR + kk * 32 + kh * 8;
            v8h a0 = *(const v8h*)(yT + abase);        // K runs of 8, contiguous
            v8h a1 = *(const v8h*)(yT + abase + 16);
            v16h a;
            #pragma unroll
            for (int i = 0; i < 8; ++i) { a[i] = a0[i]; a[i + 8] = a1[i]; }

            // B-fragment (32x16 f16): lane = col N; K = kh*8 + 0..7 and
            // K = kh*8 + 16..23 (same K order as A).
            const _Float16* bp = wq + (size_t)o * C + kk * 32 + kh * 8;
            v8h b0 = *(const v8h*)bp;
            v8h b1 = *(const v8h*)(bp + 16);
            v16h b;
            #pragma unroll
            for (int i = 0; i < 8; ++i) { b[i] = b0[i]; b[i + 8] = b1[i]; }
            if (kk + 1 < KT) __builtin_prefetch(bp + 32, 0, 1);

            acc = __builtin_amdgcn_wmma_f32_16x16x32_f16(
                false, a, false, b, (short)0, acc, false, false);
        }
        // C layout: lanes 0-15 rows M=v, lanes 16-31 rows M=v+8; col N=lm.
        #pragma unroll
        for (int v = 0; v < 8; ++v) {
            const int m = v + 8 * kh;
            const int w = w0 + m;
            if (w < W)
                out[((size_t)(nb * C) + o) * HW + h * W + w] = acc[v] * spw;
        }
    }
}

extern "C" void kernel_launch(void* const* d_in, const int* in_sizes, int n_in,
                              void* d_out, int out_size, void* d_ws, size_t ws_size,
                              hipStream_t stream) {
    const float* x    = (const float*)d_in[0];
    const float* w_dw = (const float*)d_in[1];
    const float* w_pw = (const float*)d_in[2];
    float* out = (float*)d_out;

    char* ws = (char*)d_ws;
    float*     mdw = (float*)(ws + WS_MDW);
    float*     udw = (float*)(ws + WS_UDW);
    float*     mpw = (float*)(ws + WS_MPW);
    _Float16*  wq  = (_Float16*)(ws + WS_WQ);

    hipLaunchKernelGGL(k_quant_dw, dim3(1), dim3(384), 0, stream, w_dw, mdw, udw);
    hipLaunchKernelGGL(k_pw_scale, dim3(1), dim3(256), 0, stream, w_pw, mpw);
    hipLaunchKernelGGL(k_quant_pw, dim3((C * C + 255) / 256), dim3(256), 0, stream,
                       w_pw, mpw, wq);
    hipLaunchKernelGGL(k_fused, dim3(NB * H * WT), dim3(256), 0, stream,
                       x, mdw, udw, mpw, wq, out);
}
